// Decoder_73495480369571
// MI455X (gfx1250) — matrix operand order repaired and checked by reading, not compiled, not run on hardware
//
#include <hip/hip_runtime.h>

// ---------------------------------------------------------------------------
// CDNA5 (gfx1250) implicit-GEMM 3D transposed-conv decoder.
//   * ~8.3 GFLOP total, 75MB f32 weights -> weight-bandwidth bound
//     (~1.6us HBM floor at 23.3 TB/s once weights are f16; all packed
//     weights also fit in the 192MB L2, so reuse across M-tiles is free).
//   * Weights converted once to f16 AND pre-packed into WMMA B-fragment
//     order (1KB per 32x16 fragment, lane-major): the GEMM's B feed is one
//     contiguous 32B v16h load per lane (2x global_load_b128, coalesced,
//     compile-time immediate offsets).
//   * Invalid conv-transpose taps feed A from a zeroed guard row instead of
//     a divergent branch: the inner loop is pure b128 loads + v_wmma.
//   * All matrix math: v_wmma_f32_16x16x32_f16, f32 accumulate.
//   * One wave32 per 16(M=flat spatial)x16(N=cout) output tile.
// ---------------------------------------------------------------------------

typedef _Float16 h16;
typedef __attribute__((ext_vector_type(16))) _Float16 v16h;
typedef __attribute__((ext_vector_type(8)))  _Float16 v8h;
typedef __attribute__((ext_vector_type(8)))  float    v8f;

// ---------------- zero fill (guard row) ------------------------------------
__global__ void zero_fill(h16* __restrict__ dst, int n) {
  int i = blockIdx.x * blockDim.x + threadIdx.x;
  int stride = gridDim.x * blockDim.x;
  for (; i < n; i += stride) dst[i] = (h16)0.0f;
}

// ---------------- NCDHW f32 -> channels-last f16 ---------------------------
__global__ void cvt_input(const float* __restrict__ x,
                          h16* __restrict__ dst, int C, int Sp) {
  int i = blockIdx.x * blockDim.x + threadIdx.x;
  int n = C * Sp;
  int stride = gridDim.x * blockDim.x;
  for (; i < n; i += stride) {
    int c = i % C;
    int s = i / C;
    dst[s * C + c] = (h16)x[c * Sp + s];
  }
}

// ---------------- weights: f32 DHWIO -> f16 B-fragment order ---------------
// dst fragment fid = (tap*kTiles + kc)*nTilesN + nt, 512 halves each.
// Element r = lane*16 + i of a fragment holds
//   W[tap][kc*32 + (i>=8 ? 16:0) + (lane>=16 ? 8:0) + (i&7)][nt*16 + (lane&15)]
// matching the 16-bit B-matrix (32x16) VGPR layout of wmma_f32_16x16x32_f16.
__global__ void pack_weights(const float* __restrict__ w,
                             h16* __restrict__ dst, int cin, int cout) {
  const int kTiles = cin >> 5;
  const int nTiles = cout >> 4;
  const size_t total = (size_t)27 * cin * cout;
  size_t t = (size_t)blockIdx.x * blockDim.x + threadIdx.x;
  const size_t gstride = (size_t)gridDim.x * blockDim.x;
  for (; t < total; t += gstride) {
    const int r = (int)(t & 511);
    const size_t fid = t >> 9;
    const int L = r >> 4;
    const int i = r & 15;
    const int nt = (int)(fid % nTiles);
    const size_t q = fid / nTiles;
    const int kc = (int)(q % kTiles);
    const int tap = (int)(q / kTiles);
    const int k = (kc << 5) + ((i >> 3) << 4) + ((L >> 4) << 3) + (i & 7);
    const int co = (nt << 4) + (L & 15);
    dst[t] = (h16)w[((size_t)tap * cin + k) * cout + co];
  }
}

// ---------------- implicit-GEMM conv-transpose layer (WMMA) ----------------
// X : [SIN^3][CIN]  f16 channels-last;  Wp : packed fragments;  Zg : zeros.
// Y : [Sout^3][COUT] f16 channels-last (optional ReLU).
// STRIDE==1: i = o + d - 1
// STRIDE==2: valid iff (o+d) even, i = ((o+d)>>1) - 1  (JAX SAME pads (2,1))
template <int STRIDE, int SIN, int SOSH, int CIN, int COUT>
__global__ void ct_layer_wmma(const h16* __restrict__ X,
                              const h16* __restrict__ Wp,
                              const h16* __restrict__ Zg,
                              h16* __restrict__ Y, int relu) {
  constexpr int SOUT    = 1 << SOSH;
  constexpr int SMASK   = SOUT - 1;
  constexpr int NTN     = COUT >> 4;                 // n tiles
  constexpr int NTM     = (SOUT * SOUT * SOUT) >> 4; // m tiles
  constexpr int KT      = CIN >> 5;                  // k chunks per tap
  constexpr int WSTEP   = NTN * 512;                 // halves per k-step

  const int lane = threadIdx.x & 31;
  const int wid  = blockIdx.x * (blockDim.x >> 5) + (threadIdx.x >> 5);
  if (wid >= NTM * NTN) return;  // wave-uniform exit

  const int mt = wid / NTN;
  const int nt = wid - mt * NTN;
  const int m0 = mt << 4;

  // This lane's A-matrix row (flat output spatial position)
  const int mrow = m0 + (lane & 15);
  const int ox = mrow & SMASK;
  const int oy = (mrow >> SOSH) & SMASK;
  const int oz = mrow >> (2 * SOSH);
  const int kA = (lane >> 4) << 3;  // A K-half within 32-chunk: 0 or 8

  // B walks the packed weights linearly across the whole reduction.
  const h16* wp = Wp + (size_t)nt * 512 + (size_t)lane * 16;

  v8f acc = {};

#pragma unroll
  for (int dz = 0; dz < 3; ++dz)
#pragma unroll
    for (int dy = 0; dy < 3; ++dy)
#pragma unroll
      for (int dx = 0; dx < 3; ++dx) {
        int iz, iy, ix;
        bool v;
        if (STRIDE == 2) {
          v = ((((oz + dz) | (oy + dy) | (ox + dx)) & 1) == 0);
          iz = ((oz + dz) >> 1) - 1;
          iy = ((oy + dy) >> 1) - 1;
          ix = ((ox + dx) >> 1) - 1;
        } else {
          v = true;
          iz = oz + dz - 1;
          iy = oy + dy - 1;
          ix = ox + dx - 1;
        }
        v = v && (iz >= 0) && (iz < SIN) && (iy >= 0) && (iy < SIN) &&
            (ix >= 0) && (ix < SIN);

        // Invalid lanes read the zeroed guard row: no divergent branch.
        const h16* xbase =
            v ? (X + (size_t)(((iz * SIN) + iy) * SIN + ix) * CIN + kA)
              : (Zg + kA);

#pragma unroll 4
        for (int kc = 0; kc < KT; ++kc) {
          v8h lo = *reinterpret_cast<const v8h*>(xbase + kc * 32);
          v8h hi = *reinterpret_cast<const v8h*>(xbase + kc * 32 + 16);
          v16h a = __builtin_shufflevector(lo, hi, 0, 1, 2, 3, 4, 5, 6, 7, 8,
                                           9, 10, 11, 12, 13, 14, 15);
          v16h b = *reinterpret_cast<const v16h*>(wp + (size_t)kc * WSTEP);
          acc = __builtin_amdgcn_wmma_f32_16x16x32_f16(
              false, a, false, b, (short)0, acc, false, false);
        }
        wp += (size_t)KT * WSTEP;
      }

  // ---- store D (f32 16x16 layout: N=lane&15, M = r + 8*(lane>=16)) --------
  const int nout  = (nt << 4) + (lane & 15);
  const int rbase = (lane >> 4) << 3;
#pragma unroll
  for (int r = 0; r < 8; ++r) {
    float val = acc[r];
    if (relu) val = fmaxf(val, 0.0f);
    Y[(size_t)(m0 + rbase + r) * COUT + nout] = (h16)val;
  }
}

// ---------------- final layer: 32 -> 1 channel, f32 out, no ReLU -----------
__global__ void final_layer(const h16* __restrict__ X,
                            const float* __restrict__ W6,
                            float* __restrict__ out, int S, int cin) {
  int s = blockIdx.x * blockDim.x + threadIdx.x;
  int Mtot = S * S * S;
  if (s >= Mtot) return;
  int ox = s % S;
  int t  = s / S;
  int oy = t % S;
  int oz = t / S;
  float sum = 0.0f;
  for (int dz = 0; dz < 3; ++dz) {
    int iz = oz + dz - 1;
    if (iz < 0 || iz >= S) continue;
    for (int dy = 0; dy < 3; ++dy) {
      int iy = oy + dy - 1;
      if (iy < 0 || iy >= S) continue;
      for (int dx = 0; dx < 3; ++dx) {
        int ix = ox + dx - 1;
        if (ix < 0 || ix >= S) continue;
        const h16* xr = X + (size_t)(((iz * S) + iy) * S + ix) * cin;
        const float* wr = W6 + ((dz * 3 + dy) * 3 + dx) * cin;  // cout==1
        for (int c = 0; c < cin; ++c) sum += (float)xr[c] * wr[c];
      }
    }
  }
  out[s] = sum;
}

// ---------------------------------------------------------------------------
extern "C" void kernel_launch(void* const* d_in, const int* in_sizes, int n_in,
                              void* d_out, int out_size, void* d_ws,
                              size_t ws_size, hipStream_t stream) {
  (void)in_sizes; (void)n_in; (void)out_size; (void)ws_size;

  const float* x  = (const float*)d_in[0];
  const float* w1 = (const float*)d_in[1];
  const float* w2 = (const float*)d_in[2];
  const float* w3 = (const float*)d_in[3];
  const float* w4 = (const float*)d_in[4];
  const float* w5 = (const float*)d_in[5];
  const float* w6 = (const float*)d_in[6];
  float* out = (float*)d_out;

  char* ws = (char*)d_ws;
  size_t off = 0;
  auto carve = [&](size_t bytes) -> char* {
    char* p = ws + off;
    off = (off + bytes + 255) & ~(size_t)255;
    return p;
  };

  // activations (channels-last f16) + zero guard row
  h16* Zg = (h16*)carve((size_t)1024 * 2);         // zero guard (max cin)
  h16* X0 = (h16*)carve((size_t)64 * 1024 * 2);    // 4^3  x 1024
  h16* A1 = (h16*)carve((size_t)512 * 512 * 2);    // 8^3  x 512
  h16* A2 = (h16*)carve((size_t)512 * 256 * 2);    // 8^3  x 256
  h16* A3 = (h16*)carve((size_t)4096 * 128 * 2);   // 16^3 x 128
  h16* A4 = (h16*)carve((size_t)4096 * 64 * 2);    // 16^3 x 64
  h16* A5 = (h16*)carve((size_t)32768 * 32 * 2);   // 32^3 x 32
  // packed f16 weights
  const int nw1 = 27 * 1024 * 512;
  const int nw2 = 27 * 512 * 256;
  const int nw3 = 27 * 256 * 128;
  const int nw4 = 27 * 128 * 64;
  const int nw5 = 27 * 64 * 32;
  h16* W1p = (h16*)carve((size_t)nw1 * 2);
  h16* W2p = (h16*)carve((size_t)nw2 * 2);
  h16* W3p = (h16*)carve((size_t)nw3 * 2);
  h16* W4p = (h16*)carve((size_t)nw4 * 2);
  h16* W5p = (h16*)carve((size_t)nw5 * 2);

  // ---- conversions / packing --------------------------------------------
  zero_fill<<<1, 256, 0, stream>>>(Zg, 1024);
  cvt_input<<<64, 256, 0, stream>>>(x, X0, 1024, 64);
  pack_weights<<<4096, 256, 0, stream>>>(w1, W1p, 1024, 512);
  pack_weights<<<2048, 256, 0, stream>>>(w2, W2p, 512, 256);
  pack_weights<<<512, 256, 0, stream>>>(w3, W3p, 256, 128);
  pack_weights<<<128, 256, 0, stream>>>(w4, W4p, 128, 64);
  pack_weights<<<32, 256, 0, stream>>>(w5, W5p, 64, 32);

  // ---- layers: one wave32 per 16x16 tile, 4 waves / 128-thread block -----
  auto blocks = [](int soutShift, int cout) {
    int waves = ((1 << (3 * soutShift)) >> 4) * (cout >> 4);
    return (waves + 3) / 4;
  };

  // L1: 4^3 -> 8^3, 1024 -> 512, stride 2, ReLU
  ct_layer_wmma<2, 4, 3, 1024, 512>
      <<<blocks(3, 512), 128, 0, stream>>>(X0, W1p, Zg, A1, 1);
  // L2: 8^3 -> 8^3, 512 -> 256, stride 1, ReLU
  ct_layer_wmma<1, 8, 3, 512, 256>
      <<<blocks(3, 256), 128, 0, stream>>>(A1, W2p, Zg, A2, 1);
  // L3: 8^3 -> 16^3, 256 -> 128, stride 2, ReLU
  ct_layer_wmma<2, 8, 4, 256, 128>
      <<<blocks(4, 128), 128, 0, stream>>>(A2, W3p, Zg, A3, 1);
  // L4: 16^3 -> 16^3, 128 -> 64, stride 1, ReLU
  ct_layer_wmma<1, 16, 4, 128, 64>
      <<<blocks(4, 64), 128, 0, stream>>>(A3, W4p, Zg, A4, 1);
  // L5: 16^3 -> 32^3, 64 -> 32, stride 2, ReLU
  ct_layer_wmma<2, 16, 5, 64, 32>
      <<<blocks(5, 32), 128, 0, stream>>>(A4, W5p, Zg, A5, 1);
  // L6: 32^3 -> 32^3, 32 -> 1, stride 1, no ReLU, f32 out
  final_layer<<<(32768 + 255) / 256, 256, 0, stream>>>(A5, w6, out, 32, 32);
}
